// WaveWeaverSolo_51565377356034
// MI455X (gfx1250) — compile-verified, matrix operand order checked by reading, not run on hardware
//
#include <hip/hip_runtime.h>

typedef __bf16 bf16_t;
typedef __attribute__((ext_vector_type(8)))  bf16_t v8bf;
typedef __attribute__((ext_vector_type(16))) bf16_t v16bf;
typedef __attribute__((ext_vector_type(8)))  float  v8f;

namespace {
constexpr int Vv = 32000, Ee = 512, Hh = 512, AA = 64, BB = 4, SS = 1024;
constexpr int G4 = 2048, LPYR = 3;
constexpr long WWE = (long)G4 * Hh;         // elems per 2048x512 LSTM weight matrix
}

// ---------------- device helpers ----------------

__device__ __forceinline__ float sigf(float x) { return 1.0f / (1.0f + __expf(-x)); }

// Load a 16x32 bf16 A/B fragment row for this lane.
// rowptr: base of this lane's (m or n) row in a (rows x K) row-major bf16 matrix.
// Layout per CDNA5 ISA 7.12.2 (16-bit A 16x32): lane half=l/16,
//   elems 0..7  -> k = k0 + 8*half + 0..7
//   elems 8..15 -> k = k0 + 16 + 8*half + 0..7
__device__ __forceinline__ v16bf ld_frag(const bf16_t* rowptr, int k0, int half) {
  const v8bf lo = *(const v8bf*)(rowptr + k0 + 8 * half);
  const v8bf hi = *(const v8bf*)(rowptr + k0 + 16 + 8 * half);
  return __builtin_shufflevector(lo, hi, 0, 1, 2, 3, 4, 5, 6, 7,
                                 8, 9, 10, 11, 12, 13, 14, 15);
}

#define WMMA_BF16(a, b, c) \
  __builtin_amdgcn_wmma_f32_16x16x32_bf16(false, (a), false, (b), (short)0, (c), false, false)

// ---------------- generic bf16 WMMA GEMM:  C = A @ W^T + bias0 + bias1 ----------------
// A: (M,K) bf16 row-major, W: (N,K) bf16 row-major, C: (M,N) f32.
// Requires M%16==0, K%32==0, (N/16)%4==0. Each wave computes 4 N-tiles (16x64 output),
// reusing one A fragment for 4 WMMAs. Batched via blockIdx.z with element strides.
__global__ __launch_bounds__(256) void wmma_gemm_bf16(
    const bf16_t* __restrict__ Ab, const bf16_t* __restrict__ Wb,
    const float* __restrict__ bias0, const float* __restrict__ bias1,
    float* __restrict__ Cb, int M, int N, int K, long sA, long sW, long sC) {
  const bf16_t* A = Ab + (long)blockIdx.z * sA;
  const bf16_t* W = Wb + (long)blockIdx.z * sW;
  float* C = Cb + (long)blockIdx.z * sC;

  const int wave = threadIdx.x >> 5, lane = threadIdx.x & 31;
  const int half = lane >> 4, r16 = lane & 15;
  const int tiles = N >> 4;
  const int tn0 = (blockIdx.x * 8 + wave) * 4;  // first of this wave's 4 N-tiles
  const int tm = blockIdx.y;                    // M tile index (16 rows)
  if (tn0 >= tiles) return;                     // uniform per wave -> EXEC all-ones

  const bf16_t* arow = A + (long)(tm * 16 + r16) * K;
  const bf16_t* w0 = W + (long)(tn0 * 16 + r16) * K;
  const bf16_t* w1 = w0 + (long)16 * K;
  const bf16_t* w2 = w0 + (long)32 * K;
  const bf16_t* w3 = w0 + (long)48 * K;

  v8f acc0 = {}, acc1 = {}, acc2 = {}, acc3 = {};
  for (int k0 = 0; k0 < K; k0 += 32) {
    __builtin_prefetch(w0 + k0 + 128, 0, 1);
    __builtin_prefetch(w1 + k0 + 128, 0, 1);
    __builtin_prefetch(w2 + k0 + 128, 0, 1);
    __builtin_prefetch(w3 + k0 + 128, 0, 1);
    const v16bf a = ld_frag(arow, k0, half);
    acc0 = WMMA_BF16(a, ld_frag(w0, k0, half), acc0);
    acc1 = WMMA_BF16(a, ld_frag(w1, k0, half), acc1);
    acc2 = WMMA_BF16(a, ld_frag(w2, k0, half), acc2);
    acc3 = WMMA_BF16(a, ld_frag(w3, k0, half), acc3);
  }
  v8f accs[4] = {acc0, acc1, acc2, acc3};
#pragma unroll
  for (int j = 0; j < 4; ++j) {
    const int n = (tn0 + j) * 16 + r16;
    const float badd = (bias0 ? bias0[n] : 0.0f) + (bias1 ? bias1[n] : 0.0f);
#pragma unroll
    for (int r = 0; r < 8; ++r) {
      const int m = tm * 16 + r + 8 * half;
      C[(long)m * N + n] = accs[j][r] + badd;
    }
  }
}

// ---------------- persistent LSTM scan (one block per direction) ----------------
// pre: (B,T,G4) f32 precomputed x@Wih^T + bih + bhh.  Whh: (G4,H) bf16.
// ys: (B,T,H) f32 outputs. hT: optional (B,H) final hidden (set0 only).
// Recurrent matmul done with WMMA on batch padded 4->16 (rows 4..15 zero).
__global__ __launch_bounds__(1024) void lstm_scan(
    const float* __restrict__ pre0, const bf16_t* __restrict__ Whh0,
    float* __restrict__ ys0, float* __restrict__ hT0, int rev0,
    const float* __restrict__ pre1, const bf16_t* __restrict__ Whh1,
    float* __restrict__ ys1, int rev1, int T) {
  const float* pre = pre0; const bf16_t* Whh = Whh0;
  float* ys = ys0; float* hT = hT0; int rev = rev0;
  if (blockIdx.x == 1) { pre = pre1; Whh = Whh1; ys = ys1; hT = nullptr; rev = rev1; }

  __shared__ __align__(16) bf16_t hbf[16][Hh];   // padded hidden state, bf16
  __shared__ float gacc[BB][G4];                 // gate pre-activations

  const int tid = threadIdx.x;
  for (int i = tid; i < 16 * Hh; i += 1024) ((bf16_t*)hbf)[i] = (bf16_t)0.0f;

  const int b0 = tid >> 9, u0 = tid & 511;       // this thread's two (b,u) cells
  const int b1 = b0 + 2;
  float c0 = 0.0f, c1 = 0.0f, h0v = 0.0f, h1v = 0.0f;

  const int wave = tid >> 5, lane = tid & 31;
  const int half = lane >> 4, r16 = lane & 15;

  __syncthreads();

  for (int step = 0; step < T; ++step) {
    const int t = rev ? (T - 1 - step) : step;
    // phase 1: gacc = pre[:,t,:] + h @ Whh^T
    // 32 waves x 4 tiles = 128 N-tiles; tiles processed in pairs sharing A-frag.
#pragma unroll
    for (int jp = 0; jp < 2; ++jp) {
      const int tn = wave * 4 + jp * 2;
      const bf16_t* arow = &hbf[r16][0];
      const bf16_t* w0 = Whh + (long)(tn * 16 + r16) * Hh;
      const bf16_t* w1 = w0 + (long)16 * Hh;
      v8f acc0 = {}, acc1 = {};
      for (int k0 = 0; k0 < Hh; k0 += 32) {
        const v16bf a = ld_frag(arow, k0, half);
        acc0 = WMMA_BF16(a, ld_frag(w0, k0, half), acc0);
        acc1 = WMMA_BF16(a, ld_frag(w1, k0, half), acc1);
      }
      if (half == 0) {
        const int n0 = tn * 16 + r16, n1 = n0 + 16;
#pragma unroll
        for (int r = 0; r < 4; ++r) {  // m = r (batch rows 0..3 valid)
          gacc[r][n0] = acc0[r] + pre[((long)r * T + t) * G4 + n0];
          gacc[r][n1] = acc1[r] + pre[((long)r * T + t) * G4 + n1];
        }
      }
    }
    __syncthreads();
    // phase 2: gates (torch order i,f,g,o), c kept in registers
    {
      float i0 = gacc[b0][u0],        f0 = gacc[b0][u0 + 512];
      float g0 = gacc[b0][u0 + 1024], o0 = gacc[b0][u0 + 1536];
      c0 = sigf(f0) * c0 + sigf(i0) * tanhf(g0);
      h0v = sigf(o0) * tanhf(c0);
      float i1 = gacc[b1][u0],        f1 = gacc[b1][u0 + 512];
      float g1 = gacc[b1][u0 + 1024], o1 = gacc[b1][u0 + 1536];
      c1 = sigf(f1) * c1 + sigf(i1) * tanhf(g1);
      h1v = sigf(o1) * tanhf(c1);
      ys[((long)b0 * T + t) * Hh + u0] = h0v;
      ys[((long)b1 * T + t) * Hh + u0] = h1v;
      hbf[b0][u0] = (bf16_t)h0v;
      hbf[b1][u0] = (bf16_t)h1v;
    }
    __syncthreads();
  }
  if (hT) { hT[b0 * Hh + u0] = h0v; hT[b1 * Hh + u0] = h1v; }
}

// ---------------- small fp32 dense layer (M tiny): C = A @ W^T + b ----------------
__global__ void linear_f32(const float* __restrict__ A, int lda,
                           const float* __restrict__ W, const float* __restrict__ bias,
                           float* __restrict__ C, int M, int N, int K, int relu) {
  const int i = blockIdx.x * blockDim.x + threadIdx.x;
  if (i >= M * N) return;
  const int m = i / N, n = i % N;
  const float* a = A + (long)m * lda;
  const float* w = W + (long)n * K;
  float s = bias ? bias[n] : 0.0f;
  for (int k = 0; k < K; ++k) s += a[k] * w[k];
  if (relu) s = fmaxf(s, 0.0f);
  C[(long)m * N + n] = s;
}

// ---------------- misc pointwise kernels ----------------

__global__ void f2bf_k(const float* __restrict__ in, bf16_t* __restrict__ out, long n) {
  long i = (long)blockIdx.x * blockDim.x + threadIdx.x;
  const long stride = (long)gridDim.x * blockDim.x;
  for (; i < n; i += stride) out[i] = (bf16_t)in[i];
}

__global__ void embed_k(const float* __restrict__ emb, const int* __restrict__ ids,
                        bf16_t* __restrict__ xb) {
  const int row = blockIdx.x;            // b*S + s
  const int id = ids[row];
  for (int j = threadIdx.x; j < Ee; j += blockDim.x)
    xb[(long)row * Ee + j] = (bf16_t)emb[(long)id * Ee + j];
}

__global__ void rownorm_k(const float* __restrict__ X, float* __restrict__ out, int K) {
  __shared__ float red[256];
  const int row = blockIdx.x;
  float s = 0.0f;
  for (int k = threadIdx.x; k < K; k += 256) {
    const float v = X[(long)row * K + k];
    s += v * v;
  }
  red[threadIdx.x] = s;
  __syncthreads();
  for (int st = 128; st > 0; st >>= 1) {
    if (threadIdx.x < st) red[threadIdx.x] += red[threadIdx.x + st];
    __syncthreads();
  }
  if (threadIdx.x == 0) out[row] = sqrtf(red[0]);
}

// pos/amp/phs from ACN params. params: (B, A*(H+2)) row-major.
__global__ void acn_post_k(const float* __restrict__ params, bf16_t* __restrict__ pos_bf,
                           float* __restrict__ pn, float* __restrict__ amp,
                           float* __restrict__ phs) {
  __shared__ float red[256];
  const int ba = blockIdx.x;             // b*64 + a
  const int b = ba >> 6, a = ba & 63;
  const float* row = params + (long)b * (AA * (Hh + 2)) + (long)a * (Hh + 2);
  float s = 0.0f;
  for (int j = threadIdx.x; j < Hh; j += 256) {
    const float v = row[j];
    pos_bf[(long)ba * Hh + j] = (bf16_t)v;
    s += v * v;
  }
  red[threadIdx.x] = s;
  __syncthreads();
  for (int st = 128; st > 0; st >>= 1) {
    if (threadIdx.x < st) red[threadIdx.x] += red[threadIdx.x + st];
    __syncthreads();
  }
  if (threadIdx.x == 0) {
    pn[ba] = sqrtf(red[0]);
    amp[ba] = sigf(row[Hh]);
    phs[ba] = sigf(row[Hh + 1]) * 6.28318530717958647692f;
  }
}

// wave modulation: mod = g + BETA*(enh_s*e_dir - can_s*n_dir); write cat slice 0 + bf16 copy
__global__ void wave_mod_k(const float* __restrict__ g, const float* __restrict__ num,
                           const float* __restrict__ gn, const float* __restrict__ pn,
                           const float* __restrict__ amp, const float* __restrict__ phs,
                           const float* __restrict__ e_dir, const float* __restrict__ n_dir,
                           float* __restrict__ cat, bf16_t* __restrict__ mod_bf) {
  __shared__ float wv[AA];
  __shared__ float es, cs;
  const int bs = blockIdx.x;             // b*S + s
  const int b = bs / SS;
  const int tid = threadIdx.x;
  if (tid < AA) {
    const float d = num[(long)bs * AA + tid] /
                    fmaxf(gn[bs] * pn[b * AA + tid], 1e-8f);
    wv[tid] = amp[b * AA + tid] * __cosf(d * 5.0f + phs[b * AA + tid]);
  }
  __syncthreads();
  if (tid == 0) {
    float s1 = 0.0f, s2 = 0.0f;
    for (int a = 0; a < 32; ++a) { s1 += wv[a]; s2 += wv[a + 32]; }
    es = s1; cs = s2;
  }
  __syncthreads();
  for (int h = tid; h < Hh; h += blockDim.x) {
    const float m = g[(long)bs * Hh + h] +
                    0.1f * (es * e_dir[b * Hh + h] - cs * n_dir[b * Hh + h]);
    cat[(long)bs * (4 * Hh) + h] = m;
    mod_bf[(long)bs * Hh + h] = (bf16_t)m;
  }
}

// pooled = mean_{q<4}(ys_f + ys_b); also bf16 copy for next level
__global__ void pool_k(const float* __restrict__ yf, const float* __restrict__ yb,
                       float* __restrict__ pooled, bf16_t* __restrict__ pooled_bf, int Lc) {
  const int Lp = Lc / 4;
  const long tot = (long)BB * Lp * Hh;
  const long i = (long)blockIdx.x * blockDim.x + threadIdx.x;
  if (i >= tot) return;
  const int h = (int)(i % Hh);
  const int p = (int)((i / Hh) % Lp);
  const int b = (int)(i / ((long)Hh * Lp));
  const long base = ((long)b * Lc + (long)p * 4) * Hh + h;
  float s = 0.0f;
#pragma unroll
  for (int q = 0; q < 4; ++q) s += yf[base + (long)q * Hh] + yb[base + (long)q * Hh];
  s *= 0.25f;
  pooled[i] = s;
  pooled_bf[i] = (bf16_t)s;
}

// linear interpolation (align_corners=False) of pooled (B,Lp,H) to S, into cat slice
__global__ void interp_k(const float* __restrict__ pooled, float* __restrict__ cat,
                         int Lp, int slice) {
  const long tot = (long)BB * SS * Hh;
  const long i = (long)blockIdx.x * blockDim.x + threadIdx.x;
  if (i >= tot) return;
  const int h = (int)(i % Hh);
  const int s = (int)((i / Hh) % SS);
  const int b = (int)(i / ((long)Hh * SS));
  float src = (s + 0.5f) * ((float)Lp / (float)SS) - 0.5f;
  src = fmaxf(src, 0.0f);
  int i0 = (int)floorf(src);
  if (i0 > Lp - 1) i0 = Lp - 1;
  int i1 = i0 + 1; if (i1 > Lp - 1) i1 = Lp - 1;
  const float w = src - (float)i0;
  const float v = pooled[((long)b * Lp + i0) * Hh + h] * (1.0f - w) +
                  pooled[((long)b * Lp + i1) * Hh + h] * w;
  cat[((long)b * SS + s) * (4 * Hh) + (long)slice * Hh + h] = v;
}

// ---------------- host-side orchestration ----------------

extern "C" void kernel_launch(void* const* d_in, const int* in_sizes, int n_in,
                              void* d_out, int out_size, void* d_ws, size_t ws_size,
                              hipStream_t stream) {
  (void)in_sizes; (void)n_in; (void)out_size; (void)ws_size;
  const float* emb      = (const float*)d_in[0];
  const float* gen_Wih  = (const float*)d_in[1];
  const float* gen_Whh  = (const float*)d_in[2];
  const float* gen_bih  = (const float*)d_in[3];
  const float* gen_bhh  = (const float*)d_in[4];
  const float* ctx_Wih  = (const float*)d_in[5];
  const float* ctx_Whh  = (const float*)d_in[6];
  const float* ctx_bih  = (const float*)d_in[7];
  const float* ctx_bhh  = (const float*)d_in[8];
  const float* cvg_W    = (const float*)d_in[9];
  const float* cvg_b    = (const float*)d_in[10];
  const float* acn_W1   = (const float*)d_in[11];
  const float* acn_b1   = (const float*)d_in[12];
  const float* acn_W2   = (const float*)d_in[13];
  const float* acn_b2   = (const float*)d_in[14];
  const float* proj_e_W = (const float*)d_in[15];
  const float* proj_n_W = (const float*)d_in[16];
  const float* pyr_Wih  = (const float*)d_in[17];
  const float* pyr_Whh  = (const float*)d_in[18];
  const float* pyr_bih  = (const float*)d_in[19];
  const float* pyr_bhh  = (const float*)d_in[20];
  const float* outproj_W= (const float*)d_in[21];
  const float* outproj_b= (const float*)d_in[22];
  const float* fc_W     = (const float*)d_in[23];
  const float* fc_b     = (const float*)d_in[24];
  const int*   input_ids= (const int*)d_in[25];
  float* out = (float*)d_out;

  // ---- carve workspace ----
  char* base = (char*)d_ws;
  size_t off = 0;
  auto alloc = [&](size_t bytes) -> char* {
    char* r = base + off;
    off = (off + bytes + 255) & ~(size_t)255;
    return r;
  };
  const long MBS = (long)BB * SS;  // 4096

  bf16_t* ctxWih_bf  = (bf16_t*)alloc(WWE * 2);
  bf16_t* ctxWhh_bf  = (bf16_t*)alloc(WWE * 2);
  bf16_t* genWih_bf  = (bf16_t*)alloc(2 * WWE * 2);
  bf16_t* genWhh_bf  = (bf16_t*)alloc(2 * WWE * 2);
  bf16_t* pyrWih_bf  = (bf16_t*)alloc(6 * WWE * 2);
  bf16_t* pyrWhh_bf  = (bf16_t*)alloc(6 * WWE * 2);
  bf16_t* outprojW_bf= (bf16_t*)alloc((long)Hh * (4 * Hh) * 2);
  bf16_t* fcW_bf     = (bf16_t*)alloc((long)Vv * Hh * 2);
  bf16_t* xb         = (bf16_t*)alloc(MBS * Ee * 2);
  float*  pre_f      = (float*)alloc(MBS * G4 * 4);
  float*  pre_b      = (float*)alloc(MBS * G4 * 4);
  float*  ysA        = (float*)alloc(MBS * Hh * 4);
  float*  ysB        = (float*)alloc(MBS * Hh * 4);
  bf16_t* bfA        = (bf16_t*)alloc(MBS * Hh * 2);
  bf16_t* bfB        = (bf16_t*)alloc(MBS * Hh * 2);
  float*  pyr_ysf    = (float*)alloc(MBS * Hh * 4);
  float*  pyr_ysb    = (float*)alloc(MBS * Hh * 4);
  float*  pooled     = (float*)alloc((long)BB * (SS / 4) * Hh * 4);
  bf16_t* pooled_bf  = (bf16_t*)alloc((long)BB * (SS / 4) * Hh * 2);
  float*  cat        = (float*)alloc(MBS * (4 * Hh) * 4);
  bf16_t* cat_bf     = (bf16_t*)alloc(MBS * (4 * Hh) * 2);
  float*  proj       = (float*)alloc(MBS * Hh * 4);
  bf16_t* proj_bf    = (bf16_t*)alloc(MBS * Hh * 2);
  float*  num        = (float*)alloc(MBS * AA * 4);
  float*  gn         = (float*)alloc(MBS * 4);
  float*  ctx_h      = (float*)alloc(BB * Hh * 4);
  float*  cv         = (float*)alloc(BB * 2 * Ee * 4);
  float*  h1         = (float*)alloc(BB * 128 * 4);
  float*  params     = (float*)alloc((long)BB * AA * (Hh + 2) * 4);
  float*  e_dir      = (float*)alloc(BB * Ee * 4);
  float*  n_dir      = (float*)alloc(BB * Ee * 4);
  bf16_t* pos_bf     = (bf16_t*)alloc((long)BB * AA * Hh * 2);
  float*  pn         = (float*)alloc(BB * AA * 4);
  float*  amp        = (float*)alloc(BB * AA * 4);
  float*  phs        = (float*)alloc(BB * AA * 4);

  auto cvt = [&](const float* src, bf16_t* dst, long n) {
    int blocks = (int)((n + 255) / 256);
    if (blocks > 4096) blocks = 4096;
    f2bf_k<<<blocks, 256, 0, stream>>>(src, dst, n);
  };
  // N/16 must be a multiple of 4 (true for 64, 512, 2048, 32000)
  auto gemm = [&](const bf16_t* A, const bf16_t* W, const float* b0, const float* b1,
                  float* C, int M, int N, int K, long sA, long sW, long sC, int nb) {
    wmma_gemm_bf16<<<dim3((N / 64 + 7) / 8, M / 16, nb), 256, 0, stream>>>(
        A, W, b0, b1, C, M, N, K, sA, sW, sC);
  };

  // ---- weight conversions ----
  cvt(ctx_Wih, ctxWih_bf, WWE);
  cvt(ctx_Whh, ctxWhh_bf, WWE);
  cvt(gen_Wih, genWih_bf, 2 * WWE);
  cvt(gen_Whh, genWhh_bf, 2 * WWE);
  cvt(pyr_Wih, pyrWih_bf, 6 * WWE);
  cvt(pyr_Whh, pyrWhh_bf, 6 * WWE);
  cvt(outproj_W, outprojW_bf, (long)Hh * 4 * Hh);
  cvt(fc_W, fcW_bf, (long)Vv * Hh);

  // ---- embedding ----
  embed_k<<<(int)MBS, 256, 0, stream>>>(emb, input_ids, xb);

  // ---- ctx LSTM (final hidden only) ----
  gemm(xb, ctxWih_bf, ctx_bih, ctx_bhh, pre_f, (int)MBS, G4, Ee, 0, 0, 0, 1);
  lstm_scan<<<1, 1024, 0, stream>>>(pre_f, ctxWhh_bf, ysA, ctx_h, 0,
                                    nullptr, nullptr, nullptr, 0, SS);

  // ---- control head (small, fp32 scalar) ----
  linear_f32<<<(BB * 1024 + 255) / 256, 256, 0, stream>>>(ctx_h, Hh, cvg_W, cvg_b,
                                                          cv, BB, 1024, Hh, 0);
  linear_f32<<<(BB * 128 + 255) / 256, 256, 0, stream>>>(cv, 1024, acn_W1, acn_b1,
                                                         h1, BB, 128, 1024, 1);
  linear_f32<<<(BB * AA * (Hh + 2) + 255) / 256, 256, 0, stream>>>(
      h1, 128, acn_W2, acn_b2, params, BB, AA * (Hh + 2), 128, 0);
  linear_f32<<<(BB * Hh + 255) / 256, 256, 0, stream>>>(cv, 1024, proj_e_W, nullptr,
                                                        e_dir, BB, Hh, Hh, 0);
  linear_f32<<<(BB * Hh + 255) / 256, 256, 0, stream>>>(cv + Hh, 1024, proj_n_W, nullptr,
                                                        n_dir, BB, Hh, Hh, 0);
  acn_post_k<<<BB * AA, 256, 0, stream>>>(params, pos_bf, pn, amp, phs);

  // ---- generator LSTM stack ----
  gemm(xb, genWih_bf, gen_bih, gen_bhh, pre_f, (int)MBS, G4, Ee, 0, 0, 0, 1);
  lstm_scan<<<1, 1024, 0, stream>>>(pre_f, genWhh_bf, ysB, nullptr, 0,
                                    nullptr, nullptr, nullptr, 0, SS);
  cvt(ysB, bfB, MBS * Hh);
  gemm(bfB, genWih_bf + WWE, gen_bih + G4, gen_bhh + G4, pre_f, (int)MBS, G4, Hh,
       0, 0, 0, 1);
  lstm_scan<<<1, 1024, 0, stream>>>(pre_f, genWhh_bf + WWE, ysA, nullptr, 0,
                                    nullptr, nullptr, nullptr, 0, SS);
  cvt(ysA, bfA, MBS * Hh);   // g (final generator output), bf16

  // ---- wave modulation ----
  rownorm_k<<<(int)MBS, 256, 0, stream>>>(ysA, gn, Hh);
  // num[b] = g[b] @ pos[b]^T : batched WMMA GEMM, M=1024 N=64 K=512
  gemm(bfA, pos_bf, nullptr, nullptr, num, SS, AA, Hh,
       (long)SS * Hh, (long)AA * Hh, (long)SS * AA, BB);
  wave_mod_k<<<(int)MBS, 256, 0, stream>>>(ysA, num, gn, pn, amp, phs, e_dir, n_dir,
                                           cat, bfB);  // bfB = mod (bf16)

  // ---- pyramid (3 bidirectional levels, pool by 4, interp back to S) ----
  int Lc = SS;
  const bf16_t* cur_bf = bfB;
  for (int l = 0; l < LPYR; ++l) {
    const int Mrows = BB * Lc;
    gemm(cur_bf, pyrWih_bf + (long)(l * 2 + 0) * WWE,
         pyr_bih + (long)(l * 2 + 0) * G4, pyr_bhh + (long)(l * 2 + 0) * G4,
         pre_f, Mrows, G4, Hh, 0, 0, 0, 1);
    gemm(cur_bf, pyrWih_bf + (long)(l * 2 + 1) * WWE,
         pyr_bih + (long)(l * 2 + 1) * G4, pyr_bhh + (long)(l * 2 + 1) * G4,
         pre_b, Mrows, G4, Hh, 0, 0, 0, 1);
    lstm_scan<<<2, 1024, 0, stream>>>(pre_f, pyrWhh_bf + (long)(l * 2 + 0) * WWE,
                                      pyr_ysf, nullptr, 0,
                                      pre_b, pyrWhh_bf + (long)(l * 2 + 1) * WWE,
                                      pyr_ysb, 1, Lc);
    const int Lp = Lc / 4;
    pool_k<<<(int)(((long)BB * Lp * Hh + 255) / 256), 256, 0, stream>>>(
        pyr_ysf, pyr_ysb, pooled, pooled_bf, Lc);
    interp_k<<<(int)(((long)BB * SS * Hh + 255) / 256), 256, 0, stream>>>(
        pooled, cat, Lp, l + 1);
    cur_bf = pooled_bf;
    Lc = Lp;
  }

  // ---- output projection + vocab GEMM ----
  cvt(cat, cat_bf, MBS * (4 * Hh));
  gemm(cat_bf, outprojW_bf, outproj_b, nullptr, proj, (int)MBS, Hh, 4 * Hh, 0, 0, 0, 1);
  cvt(proj, proj_bf, MBS * Hh);
  gemm(proj_bf, fcW_bf, fc_b, nullptr, out, (int)MBS, Vv, Hh, 0, 0, 0, 1);
}
